// EGNNBackbone_44521630991104
// MI455X (gfx1250) — compile-verified, matrix operand order checked by reading, not run on hardware
//
#include <hip/hip_runtime.h>
#include <hip/hip_bf16.h>
#include <cstdint>
#include <math.h>

typedef __bf16 bf16;
typedef __attribute__((ext_vector_type(16))) bf16 bf16x16;
typedef __attribute__((ext_vector_type(8)))  float f32x8;
typedef int i32x4 __attribute__((ext_vector_type(4)));
typedef __attribute__((address_space(1))) i32x4 gas_i32x4;   // global
typedef __attribute__((address_space(3))) i32x4 las_i32x4;   // LDS

#define HD 256          // hidden dim
#define ROWS 32         // edges/nodes per block (2 x 16-row WMMA tiles)
#define NTHREADS 128    // 4 waves of 32

// Async global->LDS copy path (gfx1250 GLOBAL_LOAD_ASYNC_TO_LDS_B128, ASYNCcnt)
#if defined(__has_builtin)
#if __has_builtin(__builtin_amdgcn_global_load_async_to_lds_b128) && \
    __has_builtin(__builtin_amdgcn_s_wait_asynccnt)
#define HAVE_ASYNC_LDS 1
#endif
#endif

__device__ __forceinline__ float silu_f(float x)    { return x / (1.0f + __expf(-x)); }
__device__ __forceinline__ float sigmoid_f(float x) { return 1.0f / (1.0f + __expf(-x)); }

// 16x32 bf16 WMMA fragment (A-style layout) from row-major [16][ld].
// lane 0-15: row M=lane, K = k0+0..7 (v0..v3) and k0+16..23 (v4..v7)
// lane 16-31: row M=lane-16, K = k0+8..15 and k0+24..31
__device__ __forceinline__ bf16x16 frag16(const bf16* p) {
  union { uint4 u[2]; bf16x16 v; } c;
  c.u[0] = *(const uint4*)p;
  c.u[1] = *(const uint4*)(p + 16);
  return c.v;
}
__device__ __forceinline__ const bf16* frag_ptr(const bf16* base, int ld, int k0, int lane) {
  int m  = lane & 15;
  int kh = (lane & 16) ? 8 : 0;
  return base + (size_t)m * ld + (k0 + kh);
}

#define WMMA_BF16(A, B, C) \
  __builtin_amdgcn_wmma_f32_16x16x32_bf16(false, (A), false, (B), (short)0, (C), false, false)

// ---------------------------------------------------------------------------
// Fused edge kernel: 32 edges per block (2 row-tiles share each B fragment).
// ---------------------------------------------------------------------------
__global__ __launch_bounds__(NTHREADS)
void egnn_edge_kernel(const bf16*  __restrict__ hbf,
                      const float* __restrict__ pos,
                      const int*   __restrict__ rowIdx,
                      const int*   __restrict__ colIdx,
                      const bf16*  __restrict__ ew1t,   // [256][512] bf16 (transposed)
                      const float* __restrict__ ew1f,   // [513][256] f32 (row 512)
                      const float* __restrict__ eb1,
                      const bf16*  __restrict__ ew2t,   // [256][256]
                      const float* __restrict__ eb2,
                      const bf16*  __restrict__ cw1t,   // [256][256]
                      const float* __restrict__ cb1,
                      const float* __restrict__ cw2,    // [256]
                      const float* __restrict__ cb2,    // [1]
                      const float* __restrict__ aw,     // [256] or null
                      const float* __restrict__ ab,     // [1] or null
                      float* __restrict__ agg,          // [N][256]
                      float* __restrict__ posOut,       // [N][3]
                      int E, int useAtt, int useTanh)
{
  __shared__ bf16  Arow[ROWS*HD];
  __shared__ bf16  Acol[ROWS*HD];
  __shared__ bf16  M1[ROWS*HD];
  __shared__ bf16  M2[ROWS*HD];
  __shared__ bf16  CP[ROWS*HD];
  __shared__ int   rowL[ROWS], colL[ROWS];
  __shared__ float radL[ROWS][3];
  __shared__ float rnL[ROWS];
  __shared__ float attL[ROWS];
  __shared__ float redCd[ROWS][4];
  __shared__ float redAt[ROWS][4];

  const int tid  = threadIdx.x;
  const int lane = tid & 31;
  const int wave = tid >> 5;
  const int e0   = blockIdx.x * ROWS;

  if (tid < ROWS) {
    int e = e0 + tid;
    int r = 0, c = 0;
    if (e < E) { r = rowIdx[e]; c = colIdx[e]; }
    rowL[tid] = r; colL[tid] = c;
    float dx = pos[r*3+0] - pos[c*3+0];
    float dy = pos[r*3+1] - pos[c*3+1];
    float dz = pos[r*3+2] - pos[c*3+2];
    float rn = fmaxf(sqrtf(dx*dx + dy*dy + dz*dz), 1e-8f);
    radL[tid][0] = dx; radL[tid][1] = dy; radL[tid][2] = dz;
    rnL[tid] = rn;
    attL[tid] = 1.0f;
  }
  __syncthreads();

  // gather h[row], h[col] into LDS as 16B chunks (row = 32 uint4)
  {
#if HAVE_ASYNC_LDS
    for (int i = tid; i < ROWS*(HD/8); i += NTHREADS) {
      int m = i >> 5, c = i & 31;
      const bf16* gr = hbf + ((size_t)rowL[m]*HD + c*8);
      const bf16* gc = hbf + ((size_t)colL[m]*HD + c*8);
      __builtin_amdgcn_global_load_async_to_lds_b128(
          (gas_i32x4*)gr, (las_i32x4*)&Arow[i*8], 0, 0);
      __builtin_amdgcn_global_load_async_to_lds_b128(
          (gas_i32x4*)gc, (las_i32x4*)&Acol[i*8], 0, 0);
    }
    __builtin_amdgcn_s_wait_asynccnt(0);
#else
    const uint4* hb4 = (const uint4*)hbf;          // 32 uint4 per row
    uint4* ar = (uint4*)Arow;
    uint4* ac = (uint4*)Acol;
    for (int i = tid; i < ROWS*(HD/8); i += NTHREADS) {
      int m = i >> 5, c = i & 31;
      ar[i] = hb4[(size_t)rowL[m]*(HD/8) + c];
      ac[i] = hb4[(size_t)colL[m]*(HD/8) + c];
    }
#endif
  }
  __syncthreads();

  // ---- phase 1: M1 = silu(ef @ ew1 + eb1), K=513 (512 WMMA + rank-1) ----
  for (int t = 0; t < 4; ++t) {
    int n0 = (wave*4 + t)*16;
    f32x8 acc0 = {}, acc1 = {};
#pragma unroll
    for (int k0 = 0; k0 < 512; k0 += 32) {
      const bf16* abase = (k0 < 256) ? Arow : Acol;
      int kk = k0 & 255;
      bf16x16 b  = frag16(frag_ptr(ew1t + (size_t)n0*512, 512, k0, lane));
      bf16x16 a0 = frag16(frag_ptr(abase, HD, kk, lane));
      bf16x16 a1 = frag16(frag_ptr(abase + 16*HD, HD, kk, lane));
      acc0 = WMMA_BF16(a0, b, acc0);
      acc1 = WMMA_BF16(a1, b, acc1);
    }
    int n  = n0 + (lane & 15);
    int mb = (lane & 16) ? 8 : 0;
    float bias = eb1[n];
    float w512 = ew1f[512*HD + n];
#pragma unroll
    for (int r = 0; r < 8; ++r) {
      int m0 = mb + r, m1 = m0 + 16;
      M1[m0*HD + n] = (bf16)silu_f(acc0[r] + bias + rnL[m0]*w512);
      M1[m1*HD + n] = (bf16)silu_f(acc1[r] + bias + rnL[m1]*w512);
    }
  }
  __syncthreads();

  // ---- phase 2: M2 = silu(M1 @ ew2 + eb2) ----
  for (int t = 0; t < 4; ++t) {
    int n0 = (wave*4 + t)*16;
    f32x8 acc0 = {}, acc1 = {};
#pragma unroll
    for (int k0 = 0; k0 < 256; k0 += 32) {
      bf16x16 b  = frag16(frag_ptr(ew2t + (size_t)n0*HD, HD, k0, lane));
      bf16x16 a0 = frag16(frag_ptr(M1, HD, k0, lane));
      bf16x16 a1 = frag16(frag_ptr(M1 + 16*HD, HD, k0, lane));
      acc0 = WMMA_BF16(a0, b, acc0);
      acc1 = WMMA_BF16(a1, b, acc1);
    }
    int n  = n0 + (lane & 15);
    int mb = (lane & 16) ? 8 : 0;
    float bias = eb2[n];
#pragma unroll
    for (int r = 0; r < 8; ++r) {
      M2[(mb+r)*HD + n]    = (bf16)silu_f(acc0[r] + bias);
      M2[(mb+r+16)*HD + n] = (bf16)silu_f(acc1[r] + bias);
    }
  }
  __syncthreads();

  // ---- phase 3: CP = silu(M2 @ cw1 + cb1) ----
  for (int t = 0; t < 4; ++t) {
    int n0 = (wave*4 + t)*16;
    f32x8 acc0 = {}, acc1 = {};
#pragma unroll
    for (int k0 = 0; k0 < 256; k0 += 32) {
      bf16x16 b  = frag16(frag_ptr(cw1t + (size_t)n0*HD, HD, k0, lane));
      bf16x16 a0 = frag16(frag_ptr(M2, HD, k0, lane));
      bf16x16 a1 = frag16(frag_ptr(M2 + 16*HD, HD, k0, lane));
      acc0 = WMMA_BF16(a0, b, acc0);
      acc1 = WMMA_BF16(a1, b, acc1);
    }
    int n  = n0 + (lane & 15);
    int mb = (lane & 16) ? 8 : 0;
    float bias = cb1[n];
#pragma unroll
    for (int r = 0; r < 8; ++r) {
      CP[(mb+r)*HD + n]    = (bf16)silu_f(acc0[r] + bias);
      CP[(mb+r+16)*HD + n] = (bf16)silu_f(acc1[r] + bias);
    }
  }
  __syncthreads();

  // ---- phase 4: cd / att dots (4 threads per edge, 64 elems each) ----
  {
    int m = tid >> 2, c = tid & 3;
    float scd = 0.f, sat = 0.f;
    int kb = c*64;
    for (int k = kb; k < kb+64; ++k) {
      scd += (float)CP[m*HD+k] * cw2[k];
      if (useAtt) sat += (float)M2[m*HD+k] * aw[k];
    }
    redCd[m][c] = scd; redAt[m][c] = sat;
  }
  __syncthreads();
  if (tid < ROWS) {
    float scd = 0.f, sat = 0.f;
    for (int c = 0; c < 4; ++c) { scd += redCd[tid][c]; sat += redAt[tid][c]; }
    float cd = scd + cb2[0];
    if (useTanh) cd = tanhf(cd);
    if (useAtt)  attL[tid] = sigmoid_f(sat + ab[0]);
    int e = e0 + tid;
    if (e < E) {
      float inv = 1.0f / rnL[tid];
      int r = rowL[tid];
      atomicAdd(&posOut[r*3+0], cd * radL[tid][0] * inv);
      atomicAdd(&posOut[r*3+1], cd * radL[tid][1] * inv);
      atomicAdd(&posOut[r*3+2], cd * radL[tid][2] * inv);
    }
  }
  __syncthreads();

  // ---- phase 5: agg[row] += m2 * att ----
  for (int i = tid; i < ROWS*HD; i += NTHREADS) {
    int m = i >> 8;
    int e = e0 + m;
    if (e < E) {
      int n = i & (HD-1);
      atomicAdd(&agg[(size_t)rowL[m]*HD + n], (float)M2[m*HD + n] * attL[m]);
    }
  }
}

// ---------------------------------------------------------------------------
// Fused node kernel: 32 nodes per block.
// ---------------------------------------------------------------------------
__global__ __launch_bounds__(NTHREADS)
void egnn_node_kernel(const bf16*  __restrict__ hbf,
                      const float* __restrict__ hin,
                      const float* __restrict__ agg,
                      const bf16*  __restrict__ nw1t,   // [256][512]
                      const float* __restrict__ nb1,
                      const bf16*  __restrict__ nw2t,   // [256][256]
                      const float* __restrict__ nb2,
                      const float* __restrict__ lng,
                      const float* __restrict__ lnb,
                      float* __restrict__ hout,
                      int N)
{
  __shared__ bf16  Ah[ROWS*HD];
  __shared__ bf16  Ag[ROWS*HD];
  __shared__ bf16  T1[ROWS*HD];
  __shared__ float HN[ROWS*HD];

  const int tid  = threadIdx.x;
  const int lane = tid & 31;
  const int wave = tid >> 5;
  const int n0b  = blockIdx.x * ROWS;

  {
    const uint4* hb4 = (const uint4*)hbf;
    uint4* a = (uint4*)Ah;
    for (int i = tid; i < ROWS*(HD/8); i += NTHREADS) {
      int m = i >> 5, c = i & 31;
      int node = n0b + m; if (node >= N) node = 0;
      a[i] = hb4[(size_t)node*(HD/8) + c];
    }
    const float4* ag4 = (const float4*)agg;
    for (int i = tid; i < ROWS*(HD/4); i += NTHREADS) {
      int m = i >> 6, c = i & 63;
      int node = n0b + m; if (node >= N) node = 0;
      float4 v = ag4[(size_t)node*(HD/4) + c];
      int o = m*HD + c*4;
      Ag[o+0] = (bf16)v.x; Ag[o+1] = (bf16)v.y;
      Ag[o+2] = (bf16)v.z; Ag[o+3] = (bf16)v.w;
    }
  }
  __syncthreads();

  for (int t = 0; t < 4; ++t) {
    int n0 = (wave*4 + t)*16;
    f32x8 acc0 = {}, acc1 = {};
#pragma unroll
    for (int k0 = 0; k0 < 512; k0 += 32) {
      const bf16* abase = (k0 < 256) ? Ah : Ag;
      int kk = k0 & 255;
      bf16x16 b  = frag16(frag_ptr(nw1t + (size_t)n0*512, 512, k0, lane));
      bf16x16 a0 = frag16(frag_ptr(abase, HD, kk, lane));
      bf16x16 a1 = frag16(frag_ptr(abase + 16*HD, HD, kk, lane));
      acc0 = WMMA_BF16(a0, b, acc0);
      acc1 = WMMA_BF16(a1, b, acc1);
    }
    int n  = n0 + (lane & 15);
    int mb = (lane & 16) ? 8 : 0;
    float bias = nb1[n];
#pragma unroll
    for (int r = 0; r < 8; ++r) {
      T1[(mb+r)*HD + n]    = (bf16)silu_f(acc0[r] + bias);
      T1[(mb+r+16)*HD + n] = (bf16)silu_f(acc1[r] + bias);
    }
  }
  __syncthreads();

  for (int t = 0; t < 4; ++t) {
    int n0 = (wave*4 + t)*16;
    f32x8 acc0 = {}, acc1 = {};
#pragma unroll
    for (int k0 = 0; k0 < 256; k0 += 32) {
      bf16x16 b  = frag16(frag_ptr(nw2t + (size_t)n0*HD, HD, k0, lane));
      bf16x16 a0 = frag16(frag_ptr(T1, HD, k0, lane));
      bf16x16 a1 = frag16(frag_ptr(T1 + 16*HD, HD, k0, lane));
      acc0 = WMMA_BF16(a0, b, acc0);
      acc1 = WMMA_BF16(a1, b, acc1);
    }
    int n  = n0 + (lane & 15);
    int mb = (lane & 16) ? 8 : 0;
    float bias = nb2[n];
#pragma unroll
    for (int r = 0; r < 8; ++r) {
      int m0 = mb + r, m1 = m0 + 16;
      int nd0 = n0b + m0, nd1 = n0b + m1;
      float h0 = (nd0 < N) ? hin[(size_t)nd0*HD + n] : 0.0f;
      float h1 = (nd1 < N) ? hin[(size_t)nd1*HD + n] : 0.0f;
      HN[m0*HD + n] = acc0[r] + bias + h0;
      HN[m1*HD + n] = acc1[r] + bias + h1;
    }
  }
  __syncthreads();

  if (tid < ROWS) {
    int node = n0b + tid;
    if (node < N) {
      float s = 0.f, s2 = 0.f;
      for (int k = 0; k < HD; ++k) { float x = HN[tid*HD+k]; s += x; s2 += x*x; }
      float mu  = s * (1.0f/HD);
      float var = s2 * (1.0f/HD) - mu*mu;
      float inv = rsqrtf(var + 1e-5f);
      for (int k = 0; k < HD; ++k) {
        float x = (HN[tid*HD+k] - mu) * inv;
        hout[(size_t)node*HD + k] = x * lng[k] + lnb[k];
      }
    }
  }
}

// ---------------------------------------------------------------------------
__global__ void f32_to_bf16_kernel(bf16* dst, const float* src, int n) {
  int i = blockIdx.x*blockDim.x + threadIdx.x;
  if (i < n) dst[i] = (bf16)src[i];
}
// dst[n*K + k] = src[k*Nn + n]  (transpose + downconvert; reads first K rows)
__global__ void transpose_to_bf16_kernel(bf16* dst, const float* src, int K, int Nn) {
  int i = blockIdx.x*blockDim.x + threadIdx.x;
  if (i >= K*Nn) return;
  int k = i / Nn, n = i - k*Nn;
  dst[(size_t)n*K + k] = (bf16)src[i];
}

// ---------------------------------------------------------------------------
struct LayerP {
  const float *ew1,*eb1,*ew2,*eb2,*nw1,*nb1,*nw2,*nb2,*cw1,*cb1,*cw2,*cb2,*aw,*ab,*lng,*lnb;
};

extern "C" void kernel_launch(void* const* d_in, const int* in_sizes, int n_in,
                              void* d_out, int out_size, void* d_ws, size_t ws_size,
                              hipStream_t stream) {
  const float* h0   = (const float*)d_in[0];
  const float* pos0 = (const float*)d_in[1];
  const int*   eidx = (const int*)d_in[2];
  const int N = in_sizes[0] / HD;
  const int E = in_sizes[2] / 2;
  const int* rowIdx = eidx;
  const int* colIdx = eidx + E;

  // Param extraction: detect insertion-order vs alphabetical (jax tree) order.
  LayerP P[4];
  bool ins = (in_sizes[3] == (2*HD+1)*HD);   // ew1 first => insertion order
  int base = 3;
  for (int i = 0; i < 4; ++i) {
    auto g = [&](int off){ return (const float*)d_in[base + off]; };
    if (ins) {
      P[i].ew1=g(0);  P[i].eb1=g(1);  P[i].ew2=g(2);  P[i].eb2=g(3);
      P[i].nw1=g(4);  P[i].nb1=g(5);  P[i].nw2=g(6);  P[i].nb2=g(7);
      P[i].cw1=g(8);  P[i].cb1=g(9);  P[i].cw2=g(10); P[i].cb2=g(11);
      int k = 12;
      if (i > 0) { P[i].aw=g(12); P[i].ab=g(13); k = 14; }
      else       { P[i].aw=nullptr; P[i].ab=nullptr; }
      P[i].lng=g(k); P[i].lnb=g(k+1);
      base += k + 2;
    } else {
      int o = 0;
      if (i > 0) { P[i].ab=g(0); P[i].aw=g(1); o = 2; }
      else       { P[i].aw=nullptr; P[i].ab=nullptr; }
      P[i].cb1=g(o+0);  P[i].cb2=g(o+1);  P[i].cw1=g(o+2);  P[i].cw2=g(o+3);
      P[i].eb1=g(o+4);  P[i].eb2=g(o+5);  P[i].ew1=g(o+6);  P[i].ew2=g(o+7);
      P[i].lnb=g(o+8);  P[i].lng=g(o+9);  P[i].nb1=g(o+10); P[i].nb2=g(o+11);
      P[i].nw1=g(o+12); P[i].nw2=g(o+13);
      base += o + 14;
    }
  }

  // Workspace carve
  uintptr_t ws = ((uintptr_t)d_ws + 255) & ~(uintptr_t)255;
  auto carve = [&](size_t bytes) -> void* {
    void* p = (void*)ws;
    ws = (ws + bytes + 255) & ~(uintptr_t)255;
    return p;
  };
  bf16*  hbf  = (bf16*)carve((size_t)N*HD*sizeof(bf16));
  float* hA   = (float*)carve((size_t)N*HD*sizeof(float));
  float* hB   = (float*)carve((size_t)N*HD*sizeof(float));
  float* posA = (float*)carve((size_t)N*3*sizeof(float));
  float* posB = (float*)carve((size_t)N*3*sizeof(float));
  float* agg  = (float*)carve((size_t)N*HD*sizeof(float));
  bf16 *ew1t[4], *ew2t[4], *nw1t[4], *nw2t[4], *cw1t[4];
  for (int i = 0; i < 4; ++i) {
    ew1t[i] = (bf16*)carve((size_t)HD*512*sizeof(bf16));
    ew2t[i] = (bf16*)carve((size_t)HD*HD*sizeof(bf16));
    nw1t[i] = (bf16*)carve((size_t)HD*512*sizeof(bf16));
    nw2t[i] = (bf16*)carve((size_t)HD*HD*sizeof(bf16));
    cw1t[i] = (bf16*)carve((size_t)HD*HD*sizeof(bf16));
  }

  // Weight prep (bf16 transposed copies; ~4 MB total, L2-resident)
  const int TB = 256;
  for (int i = 0; i < 4; ++i) {
    transpose_to_bf16_kernel<<<(512*HD+TB-1)/TB, TB, 0, stream>>>(ew1t[i], P[i].ew1, 512, HD);
    transpose_to_bf16_kernel<<<(HD*HD+TB-1)/TB,  TB, 0, stream>>>(ew2t[i], P[i].ew2, HD, HD);
    transpose_to_bf16_kernel<<<(512*HD+TB-1)/TB, TB, 0, stream>>>(nw1t[i], P[i].nw1, 512, HD);
    transpose_to_bf16_kernel<<<(HD*HD+TB-1)/TB,  TB, 0, stream>>>(nw2t[i], P[i].nw2, HD, HD);
    transpose_to_bf16_kernel<<<(HD*HD+TB-1)/TB,  TB, 0, stream>>>(cw1t[i], P[i].cw1, HD, HD);
  }

  float* houts[4] = { hA, hB, hA, (float*)d_out };
  float* pouts[4] = { posA, posB, posA, (float*)d_out + (size_t)N*HD };

  const float* hin = h0;
  const float* pin = pos0;
  const int Eb = (E + ROWS - 1) / ROWS;
  const int Nb = (N + ROWS - 1) / ROWS;

  for (int i = 0; i < 4; ++i) {
    f32_to_bf16_kernel<<<(N*HD+TB-1)/TB, TB, 0, stream>>>(hbf, hin, N*HD);
    (void)hipMemsetAsync(agg, 0, (size_t)N*HD*sizeof(float), stream);
    (void)hipMemcpyAsync(pouts[i], pin, (size_t)N*3*sizeof(float), hipMemcpyDeviceToDevice, stream);

    egnn_edge_kernel<<<Eb, NTHREADS, 0, stream>>>(
        hbf, pin, rowIdx, colIdx,
        ew1t[i], P[i].ew1, P[i].eb1,
        ew2t[i], P[i].eb2,
        cw1t[i], P[i].cb1, P[i].cw2, P[i].cb2,
        P[i].aw, P[i].ab,
        agg, pouts[i], E, (i > 0) ? 1 : 0, (i == 3) ? 1 : 0);

    egnn_node_kernel<<<Nb, NTHREADS, 0, stream>>>(
        hbf, hin, agg,
        nw1t[i], P[i].nb1, nw2t[i], P[i].nb2,
        P[i].lng, P[i].lnb, houts[i], N);

    hin = houts[i];
    pin = pouts[i];
  }
}
